// MultiHeadAttention_15496242004510
// MI455X (gfx1250) — matrix operand order repaired and checked by reading, not compiled, not run on hardware
//
#include <hip/hip_runtime.h>

typedef unsigned short u16;
typedef unsigned int   u32;
typedef __attribute__((ext_vector_type(16))) __bf16 bf16x16;
typedef __attribute__((ext_vector_type(8)))  float  f32x8;
typedef __attribute__((ext_vector_type(4)))  float  f32x4;
typedef __attribute__((ext_vector_type(4)))  u32    u32x4;
typedef __attribute__((ext_vector_type(2)))  u32    u32x2;

// -------- constants for this problem --------
#define BATCH 2
#define SEQ   2048
#define DMODEL 1024
#define NHEAD 16
#define HDIM  64
#define NTOK  (BATCH * SEQ)        // 4096

// -------- fragment container (32 bytes = 16 bf16 per lane) --------
union FragB {
  bf16x16 v;
  u32x4   q[2];
  u16     h[16];
};

__device__ __forceinline__ u16 f2bf(float f) {
  union { float f; u32 u; } c; c.f = f;
  u32 u = c.u;
  u += 0x7FFFu + ((u >> 16) & 1u);   // round-to-nearest-even
  return (u16)(u >> 16);
}

// 16 contiguous bf16 (32B): B-operand fragment (lane = column, K contiguous)
__device__ __forceinline__ void load_contig16(FragB& f, const u16* p) {
  const u32x4* p4 = reinterpret_cast<const u32x4*>(p);
  f.q[0] = p4[0];
  f.q[1] = p4[1];
}

// A-operand fragment: two 8-element K-chunks at kbase and kbase+8
__device__ __forceinline__ void load_a_frag(FragB& f, const u16* row, int kbase) {
  f.q[0] = *reinterpret_cast<const u32x4*>(row + kbase);
  f.q[1] = *reinterpret_cast<const u32x4*>(row + kbase + 8);
}

__device__ __forceinline__ f32x8 wmma_bf16(const FragB& a, const FragB& b, f32x8 c) {
  return __builtin_amdgcn_wmma_f32_16x16x32_bf16(
      /*neg_a=*/false, a.v, /*neg_b=*/false, b.v,
      /*c_mod=*/(short)0, c, /*reuse_a=*/false, /*reuse_b=*/false);
}

// ======================= f32 -> bf16 convert (4-wide) =======================
__global__ void cvt_f32_to_bf16(const float* __restrict__ in, u16* __restrict__ out, int n4) {
  int i = blockIdx.x * blockDim.x + threadIdx.x;
  if (i < n4) {
    f32x4 v = reinterpret_cast<const f32x4*>(in)[i];
    u32x2 o;
    o[0] = (u32)f2bf(v[0]) | ((u32)f2bf(v[1]) << 16);
    o[1] = (u32)f2bf(v[2]) | ((u32)f2bf(v[3]) << 16);
    reinterpret_cast<u32x2*>(out)[i] = o;
  }
}

// ======================= GEMM: C[m,n] = sum_k A[m,k] * W[n,k] =======================
// A: [M,1024] bf16 row-major, W: [1024,1024] bf16 row-major (nn.Linear weight).
// mode 0: store bf16 to [B,H,S,Hd]   (Q, K)
// mode 1: store bf16 to [B,H,Hd,S]   (V transposed)
// mode 2: store f32  to [M,1024]     (output projection -> d_out)
// block = 256 (8 waves as 4M x 2N); wave -> 32x128 strip (16 accumulators);
// k-loop double-buffered (2 x 32). amdgpu_waves_per_eu(2) -> 512-VGPR budget so
// both pipeline stages (2 A-frags + 8 B-frags each) stay live in registers.
__global__ __launch_bounds__(256) __attribute__((amdgpu_waves_per_eu(2, 8)))
void gemm1024(const u16* __restrict__ A, const u16* __restrict__ W,
              u16* __restrict__ outb, float* __restrict__ outf, int mode) {
  const int lane = threadIdx.x & 31;
  const int wave = threadIdx.x >> 5;
  const int hl   = lane >> 4;     // which 16-lane half
  const int lr   = lane & 15;

  const int waveM = wave & 3;
  const int waveN = wave >> 2;
  const int m0 = (blockIdx.x * 4 + waveM) * 32;
  const int n0 = (blockIdx.y * 2 + waveN) * 128;

  const f32x8 zero = {0.f, 0.f, 0.f, 0.f, 0.f, 0.f, 0.f, 0.f};
  f32x8 acc[2][8];
#pragma unroll
  for (int r = 0; r < 2; ++r)
#pragma unroll
    for (int t = 0; t < 8; ++t) acc[r][t] = zero;

  // A: lane holds row; 16 contiguous K at k+hl*16
  const u16* arow0 = A + (size_t)(m0 + lr) * DMODEL + (hl ? 16 : 0);
  const u16* arow1 = arow0 + (size_t)16 * DMODEL;
  // B: lane holds column; 16 contiguous K at k+hl*16 (same k order as A)
  const u16* wcol  = W + (size_t)(n0 + lr) * DMODEL + (hl ? 16 : 0);

  FragB a0[2], b0[8];   // pipeline stage 0
  FragB a1[2], b1[8];   // pipeline stage 1

  // prologue: stage 0 holds k = 0
  load_a_frag(a0[0], arow0, 0);
  load_a_frag(a0[1], arow1, 0);
#pragma unroll
  for (int t = 0; t < 8; ++t) load_contig16(b0[t], wcol + (size_t)t * 16 * DMODEL);

  for (int k = 0; k < DMODEL; k += 64) {
    __builtin_prefetch(arow0 + k + 256, 0, 3);
    // issue stage-1 loads (k+32) while stage-0 WMMAs run
    const int k1 = k + 32;
    load_a_frag(a1[0], arow0, k1);
    load_a_frag(a1[1], arow1, k1);
#pragma unroll
    for (int t = 0; t < 8; ++t) load_contig16(b1[t], wcol + (size_t)t * 16 * DMODEL + k1);

    // consume stage 0 (k)
#pragma unroll
    for (int t = 0; t < 8; ++t) {
      acc[0][t] = wmma_bf16(a0[0], b0[t], acc[0][t]);
      acc[1][t] = wmma_bf16(a0[1], b0[t], acc[1][t]);
    }

    // issue stage-0 loads (k+64); clamp dead tail load in-bounds
    const int k2 = (k + 64 < DMODEL) ? (k + 64) : 0;
    load_a_frag(a0[0], arow0, k2);
    load_a_frag(a0[1], arow1, k2);
#pragma unroll
    for (int t = 0; t < 8; ++t) load_contig16(b0[t], wcol + (size_t)t * 16 * DMODEL + k2);

    // consume stage 1 (k+32)
#pragma unroll
    for (int t = 0; t < 8; ++t) {
      acc[0][t] = wmma_bf16(a1[0], b1[t], acc[0][t]);
      acc[1][t] = wmma_bf16(a1[1], b1[t], acc[1][t]);
    }
  }

  // C/D layout: vgpr i -> row (i + hl*8), col = lr within the 16-wide tile
#pragma unroll
  for (int r = 0; r < 2; ++r) {
#pragma unroll
    for (int t = 0; t < 8; ++t) {
#pragma unroll
      for (int i = 0; i < 8; ++i) {
        const int row = m0 + r * 16 + i + hl * 8;   // token index
        const int col = n0 + t * 16 + lr;           // feature index
        const float val = acc[r][t][i];
        if (mode == 2) {
          outf[(size_t)row * DMODEL + col] = val;
        } else {
          const int b = row >> 11;                  // / SEQ
          const int s = row & (SEQ - 1);
          const int h = col >> 6;                   // / HDIM
          const int d = col & (HDIM - 1);
          const u16 bv = f2bf(val);
          if (mode == 0)
            outb[(((size_t)(b * NHEAD + h) * SEQ + s) * HDIM) + d] = bv;
          else
            outb[(((size_t)(b * NHEAD + h) * HDIM + d) * SEQ) + s] = bv;
        }
      }
    }
  }
}

// ======================= Flash attention (one wave per 16-query tile) =======================
// Q,K: [B,H,S,Hd] bf16; VT: [B,H,Hd,S] bf16; O: [B,S,H*Hd] bf16.
// block = 128 (4 waves), grid = B*H*(S/16)/4.
__global__ __launch_bounds__(128) __attribute__((amdgpu_waves_per_eu(2, 8)))
void attn_fwd(const u16* __restrict__ Q, const u16* __restrict__ K,
              const u16* __restrict__ VT, u16* __restrict__ O,
              const int* __restrict__ causalPtr) {
  __shared__ __align__(16) u16 plds[4][16 * 32];   // P tile staging per wave

  const int lane = threadIdx.x & 31;
  const int wave = threadIdx.x >> 5;
  const int hl   = lane >> 4;
  const int lr   = lane & 15;

  int tile = blockIdx.x * 4 + wave;               // 0 .. B*H*(S/16)-1
  const int S16 = SEQ / 16;
  const int qt = tile % S16; tile /= S16;
  const int h  = tile % NHEAD; tile /= NHEAD;
  const int b  = tile;
  const int q0 = qt * 16;
  const bool causal = (*causalPtr) != 0;

  const u16* qbase = Q  + (size_t)(b * NHEAD + h) * SEQ * HDIM;
  const u16* kbase = K  + (size_t)(b * NHEAD + h) * SEQ * HDIM;
  const u16* vbase = VT + (size_t)(b * NHEAD + h) * HDIM * SEQ;
  u16* myp = &plds[wave][0];

  // Q fragments: rows = queries (lane lr), two d-slices 0..31 / 32..63
  FragB qa[2];
  const u16* qrow = qbase + (size_t)(q0 + lr) * HDIM + (hl ? 16 : 0);
  load_a_frag(qa[0], qrow, 0);
  load_a_frag(qa[1], qrow, 32);

  const f32x8 zero = {0.f, 0.f, 0.f, 0.f, 0.f, 0.f, 0.f, 0.f};
  f32x8 oacc[4];
#pragma unroll
  for (int t = 0; t < 4; ++t) oacc[t] = zero;
  float m_run[8], l_run[8];
#pragma unroll
  for (int i = 0; i < 8; ++i) { m_run[i] = -__builtin_inff(); l_run[i] = 0.f; }

  const int kend = causal ? (q0 + 16) : SEQ;
  const float scale = 0.125f;                     // 1/sqrt(64)

  for (int k0 = 0; k0 < kend; k0 += 32) {
    // second 16-key subtile may be past the causal frontier: clamp address, mask handles it
    const int kb1 = (k0 + 16 < kend) ? (k0 + 16) : k0;

    // ---- issue ALL loads for this step first: K frags, then V frags.
    //      V loads stay in flight through score WMMAs + softmax + LDS transpose.
    FragB kf[4], vf[4];
    {
      const u16* krow0 = kbase + (size_t)(k0  + lr) * HDIM + (hl ? 16 : 0);
      const u16* krow1 = kbase + (size_t)(kb1 + lr) * HDIM + (hl ? 16 : 0);
      load_contig16(kf[0], krow0);
      load_contig16(kf[1], krow0 + 32);
      load_contig16(kf[2], krow1);
      load_contig16(kf[3], krow1 + 32);
    }
#pragma unroll
    for (int t = 0; t < 4; ++t)
      load_contig16(vf[t], vbase + (size_t)(t * 16 + lr) * SEQ + k0 + (hl ? 16 : 0));

    // ---- scores: S = Q @ K^T ----
    f32x8 s0 = zero, s1 = zero;
    s0 = wmma_bf16(qa[0], kf[0], s0);
    s0 = wmma_bf16(qa[1], kf[1], s0);
    s1 = wmma_bf16(qa[0], kf[2], s1);
    s1 = wmma_bf16(qa[1], kf[3], s1);

    // ---- online softmax over the 32 keys of this step ----
#pragma unroll
    for (int i = 0; i < 8; ++i) {
      float v0 = s0[i] * scale;
      float v1 = s1[i] * scale;
      if (causal) {
        const int qi = q0 + i + hl * 8;
        if (k0 + lr      > qi) v0 = -__builtin_inff();
        if (k0 + 16 + lr > qi) v1 = -__builtin_inff();
      }
      float mx = fmaxf(v0, v1);
#pragma unroll
      for (int m = 1; m < 16; m <<= 1) mx = fmaxf(mx, __shfl_xor(mx, m, 32));
      const float mnew = fmaxf(m_run[i], mx);
      const float corr = __expf(m_run[i] - mnew);
      const float p0 = __expf(v0 - mnew);
      const float p1 = __expf(v1 - mnew);
      float rs = p0 + p1;
#pragma unroll
      for (int m = 1; m < 16; m <<= 1) rs += __shfl_xor(rs, m, 32);
      l_run[i] = l_run[i] * corr + rs;
      m_run[i] = mnew;
      oacc[0][i] *= corr; oacc[1][i] *= corr; oacc[2][i] *= corr; oacc[3][i] *= corr;
      // stage P (C-layout -> LDS row-major 16x32)
      const int prow = i + hl * 8;
      myp[prow * 32 + lr]      = f2bf(p0);
      myp[prow * 32 + 16 + lr] = f2bf(p1);
    }

    // ---- P (16x32, A-layout from LDS) @ V (32 keys x 64 dims) ----
    FragB pa;
    load_a_frag(pa, myp + lr * 32, hl ? 16 : 0);
#pragma unroll
    for (int t = 0; t < 4; ++t)
      oacc[t] = wmma_bf16(pa, vf[t], oacc[t]);
  }

  // ---- finalize: O / l, store bf16 [B,S,D] ----
#pragma unroll
  for (int i = 0; i < 8; ++i) {
    const float inv = 1.0f / l_run[i];
    const int qi = q0 + i + hl * 8;
    u16* orow = O + ((size_t)(b * SEQ + qi) * DMODEL) + h * HDIM;
#pragma unroll
    for (int t = 0; t < 4; ++t)
      orow[t * 16 + lr] = f2bf(oacc[t][i] * inv);
  }
}

// ======================= host side =======================
extern "C" void kernel_launch(void* const* d_in, const int* in_sizes, int n_in,
                              void* d_out, int out_size, void* d_ws, size_t ws_size,
                              hipStream_t stream) {
  (void)in_sizes; (void)n_in; (void)out_size; (void)ws_size;

  const float* x  = (const float*)d_in[0];
  const float* Wq = (const float*)d_in[1];
  const float* Wk = (const float*)d_in[2];
  const float* Wv = (const float*)d_in[3];
  const float* Wo = (const float*)d_in[4];
  const int* causal = (const int*)d_in[5];

  // workspace layout (bf16 elements), total 48 MiB
  u16* ws  = (u16*)d_ws;
  u16* xb  = ws;                               // 4096*1024
  u16* wqb = xb  + (size_t)NTOK * DMODEL;
  u16* wkb = wqb + (size_t)DMODEL * DMODEL;
  u16* wvb = wkb + (size_t)DMODEL * DMODEL;
  u16* wob = wvb + (size_t)DMODEL * DMODEL;
  u16* qb  = wob + (size_t)DMODEL * DMODEL;    // [B,H,S,Hd]
  u16* kb  = qb  + (size_t)NTOK * DMODEL;      // [B,H,S,Hd]
  u16* vtb = kb  + (size_t)NTOK * DMODEL;      // [B,H,Hd,S]
  u16* ob  = vtb + (size_t)NTOK * DMODEL;      // [B,S,D]

  // f32 -> bf16
  {
    const int nx4 = NTOK * DMODEL / 4;
    const int nw4 = DMODEL * DMODEL / 4;
    cvt_f32_to_bf16<<<(nx4 + 255) / 256, 256, 0, stream>>>(x,  xb,  nx4);
    cvt_f32_to_bf16<<<(nw4 + 255) / 256, 256, 0, stream>>>(Wq, wqb, nw4);
    cvt_f32_to_bf16<<<(nw4 + 255) / 256, 256, 0, stream>>>(Wk, wkb, nw4);
    cvt_f32_to_bf16<<<(nw4 + 255) / 256, 256, 0, stream>>>(Wv, wvb, nw4);
    cvt_f32_to_bf16<<<(nw4 + 255) / 256, 256, 0, stream>>>(Wo, wob, nw4);
  }

  // projections: block = 128x256 of C, wave = 32x128
  dim3 ggrid(NTOK / 128, DMODEL / 256);
  gemm1024<<<ggrid, 256, 0, stream>>>(xb, wqb, qb,  nullptr, 0);   // Q
  gemm1024<<<ggrid, 256, 0, stream>>>(xb, wkb, kb,  nullptr, 0);   // K
  gemm1024<<<ggrid, 256, 0, stream>>>(xb, wvb, vtb, nullptr, 1);   // V^T

  // attention: B*H*(S/16) = 4096 query tiles, 4 waves / block
  attn_fwd<<<(BATCH * NHEAD * (SEQ / 16)) / 4, 128, 0, stream>>>(qb, kb, vtb, ob, causal);

  // output projection -> f32 d_out
  gemm1024<<<ggrid, 256, 0, stream>>>(ob, wob, nullptr, (float*)d_out, 2);
}
